// Molecule_3393024164449
// MI455X (gfx1250) — matrix operand order repaired and checked by reading, not compiled, not run on hardware
//
#include <hip/hip_runtime.h>
#include <hip/hip_bf16.h>
#include <math.h>

typedef __attribute__((ext_vector_type(16))) _Float16 v16h;
typedef __attribute__((ext_vector_type(8)))  float    v8f;
typedef unsigned int v4u __attribute__((ext_vector_type(4)));
typedef int          v8i __attribute__((ext_vector_type(8)));
typedef int          v4i __attribute__((ext_vector_type(4)));

#define WMMA16(a, b, c) \
    __builtin_amdgcn_wmma_f32_16x16x32_f16(false, (a), false, (b), (short)0, (c), false, false)

// Use the Tensor Data Mover path only when the toolchain exposes the builtins
// (device pass). Fallback = direct global loads of the weight tile.
#if defined(__gfx1250__) && defined(__has_builtin)
#if __has_builtin(__builtin_amdgcn_tensor_load_to_lds) && \
    __has_builtin(__builtin_amdgcn_s_wait_tensorcnt)
#define USE_TDM 1
#endif
#endif
#ifndef USE_TDM
#define USE_TDM 0
#endif

// ---------------------------------------------------------------------------
// fragment loaders (fp32 memory -> f16 fragments, contiguous k = 16*h + s at all call sites)
// ---------------------------------------------------------------------------
__device__ __forceinline__ v16h cvt16(const float* __restrict__ p) {
    v16h f;
#pragma unroll
    for (int i = 0; i < 16; ++i) f[i] = (_Float16)p[i];
    return f;
}

__device__ __forceinline__ v16h cvt8x2(const float* __restrict__ a,
                                       const float* __restrict__ b) {
    v16h f;
#pragma unroll
    for (int i = 0; i < 8; ++i) { f[i] = (_Float16)a[i]; f[i + 8] = (_Float16)b[i]; }
    return f;
}

__device__ __forceinline__ v16h cvt8x2s(const float* __restrict__ a,
                                        const float* __restrict__ b, float s) {
    v16h f;
#pragma unroll
    for (int i = 0; i < 8; ++i) {
        f[i]     = (_Float16)(a[i] * s);
        f[i + 8] = (_Float16)(b[i] * s);
    }
    return f;
}

#if USE_TDM
// ---------------------------------------------------------------------------
// TDM: DMA a 2D fp32 tile (tile_dim0=32 along K, tile_dim1=128 rows, row
// stride = K elements) from global into LDS. D# per CDNA5 ISA 08 sec 8.3/8.4.
// This toolchain exposes the 6-arg builtin: (g0, g1, g2, g3, g4, cpol);
// groups 2/3 (and the trailing group) are unused for 2D tensors -> zeros.
// ---------------------------------------------------------------------------
__device__ __forceinline__ void tdm_load_w_tile(unsigned int lds_off,
                                                const float* gptr, int K, int N) {
    const unsigned long long ga = (unsigned long long)(uintptr_t)gptr;
    v4u g0;
    g0[0] = 1u;                                   // count=1 (valid user D#)
    g0[1] = lds_off;                              // lds_addr (bytes)
    g0[2] = (unsigned int)ga;                     // global_addr[31:0]
    g0[3] = (unsigned int)((ga >> 32) & 0x1FFFFFFu) | (2u << 30);  // addr[56:32] | type=2
    v8i g1;
    g1[0] = (int)(2u << 16);                      // workgroup_mask=0, data_size=2 (4B)
    g1[1] = (int)(((unsigned)K & 0xFFFFu) << 16); // tensor_dim0[15:0]
    g1[2] = (int)((((unsigned)K >> 16) & 0xFFFFu) | (((unsigned)N & 0xFFFFu) << 16));
    g1[3] = (int)((((unsigned)N >> 16) & 0xFFFFu) | (32u << 16));  // tile_dim0=32
    g1[4] = (int)128u;                            // tile_dim1=128, tile_dim2=0
    g1[5] = (int)(unsigned)K;                     // tensor_dim0_stride[31:0]
    g1[6] = 0;                                    // stride[47:32]=0, dim1_stride lo=0
    g1[7] = 0;
    v4i gz4;
    gz4[0] = gz4[1] = gz4[2] = gz4[3] = 0;        // groups 2/3 unused (2D tensor)
    v8i gz8;
#pragma unroll
    for (int i = 0; i < 8; ++i) gz8[i] = 0;
    __builtin_amdgcn_tensor_load_to_lds(g0, g1, gz4, gz4, gz8, 0);
}
#endif

// ---------------------------------------------------------------------------
// Generic WMMA GEMM:  C[M,N] = op( A[M,K] @ W[N,K]^T )  (+ R | += C, * rowscale)
// block = 256 threads (8 waves), block tile 128x128, wave tile 64x32 (4x2 WMMA)
// Weight tile (128x32 fp32) staged in LDS by the Tensor Data Mover, double
// buffered and overlapped with WMMA compute. Requires M%128==0, N%128==0,
// K%32==0.
// ---------------------------------------------------------------------------
template <int RELU, int ACC, int SCALE>
__global__ void __launch_bounds__(256)
gemm_wmma_kernel(const float* __restrict__ A, const float* __restrict__ W,
                 float* __restrict__ C, const float* __restrict__ R,
                 const float* __restrict__ rs, int rs_stride,
                 int M, int N, int K) {
    const int lane = threadIdx.x & 31;
    const int wid  = threadIdx.x >> 5;
    const int wm   = wid & 1;       // 2 wave rows  (64 rows each)
    const int wn   = wid >> 1;      // 4 wave cols  (32 cols each)
    const int r    = lane & 15;
    const int hl   = lane >> 4;

    const int m0 = blockIdx.y * 128 + wm * 64;
    const int n0 = blockIdx.x * 128 + wn * 32;

    v8f acc[4][2];
#pragma unroll
    for (int i = 0; i < 4; ++i)
#pragma unroll
        for (int j = 0; j < 2; ++j)
#pragma unroll
            for (int e = 0; e < 8; ++e) acc[i][j][e] = 0.0f;

    const float* Arow[4];
#pragma unroll
    for (int i = 0; i < 4; ++i) Arow[i] = A + (size_t)(m0 + 16 * i + r) * K;

#if USE_TDM
    __shared__ float wtile[2][128 * 32];          // 2 x 16KB double buffer
    const float* Wblk = W + (size_t)blockIdx.x * 128 * K;   // this block's 128 rows
    const unsigned int ldsoff[2] = {
        (unsigned int)(uintptr_t)&wtile[0][0],
        (unsigned int)(uintptr_t)&wtile[1][0]
    };
    if (wid == 0) tdm_load_w_tile(ldsoff[0], Wblk, K, N);
    __builtin_amdgcn_s_wait_tensorcnt(0);
    __syncthreads();

    int cur = 0;
    for (int k0 = 0; k0 < K; k0 += 32) {
        const int nxt = cur ^ 1;
        if (wid == 0 && (k0 + 32) < K)
            tdm_load_w_tile(ldsoff[nxt], Wblk + k0 + 32, K, N);

        const float* wbuf = &wtile[cur][0];
        v16h bf[2];
#pragma unroll
        for (int j = 0; j < 2; ++j)
            bf[j] = cvt16(wbuf + ((wn * 32 + j * 16 + r) << 5) + hl * 16);
#pragma unroll
        for (int i = 0; i < 4; ++i) {
            v16h af = cvt16(Arow[i] + k0 + hl * 16);
            acc[i][0] = WMMA16(af, bf[0], acc[i][0]);
            acc[i][1] = WMMA16(af, bf[1], acc[i][1]);
        }

        __builtin_amdgcn_s_wait_tensorcnt(0);     // wave0 drains TDM; others no-op
        __syncthreads();                          // publish next buffer / retire cur
        cur = nxt;
    }
#else
    const float* Wrow[2];
#pragma unroll
    for (int j = 0; j < 2; ++j) Wrow[j] = W + (size_t)(n0 + 16 * j + r) * K;

    for (int k0 = 0; k0 < K; k0 += 32) {
        v16h bf[2];
#pragma unroll
        for (int j = 0; j < 2; ++j) bf[j] = cvt16(Wrow[j] + k0 + hl * 16);
#pragma unroll
        for (int i = 0; i < 4; ++i) {
            v16h af = cvt16(Arow[i] + k0 + hl * 16);
            acc[i][0] = WMMA16(af, bf[0], acc[i][0]);
            acc[i][1] = WMMA16(af, bf[1], acc[i][1]);
        }
    }
#endif

#pragma unroll
    for (int i = 0; i < 4; ++i)
#pragma unroll
        for (int j = 0; j < 2; ++j)
#pragma unroll
            for (int e = 0; e < 8; ++e) {
                const int gm = m0 + 16 * i + e + 8 * hl;
                const int gn = n0 + 16 * j + r;
                float v = acc[i][j][e];
                if (RELU) v = fmaxf(v, 0.0f);
                if (SCALE) v *= rs[(size_t)gm * rs_stride];
                const size_t off = (size_t)gm * N + gn;
                if (ACC) v += C[off];
                else if (R) v += R[off];
                C[off] = v;
            }
}

// ---------------------------------------------------------------------------
// Flash attention: one wave per 16-row query tile, online softmax, WMMA for
// both Q@K^T and P@V.  HD must be 64.  4 waves per 128-thread block.
// ---------------------------------------------------------------------------
#define ATT_WAVES 4
__global__ void __launch_bounds__(ATT_WAVES * 32)
flash_attn_kernel(const float* __restrict__ q, const float* __restrict__ k,
                  const float* __restrict__ v, float* __restrict__ o,
                  int T, int D, int H, float scale) {
    __shared__ float    Sld[ATT_WAVES][16 * 32];
    __shared__ _Float16 Pld[ATT_WAVES][16 * 32];
    __shared__ float    mrow[ATT_WAVES][16];
    __shared__ float    lrow[ATT_WAVES][16];
    __shared__ float    arow[ATT_WAVES][16];

    const int lane = threadIdx.x & 31;
    const int wid  = threadIdx.x >> 5;
    const int r    = lane & 15;
    const int hl   = lane >> 4;

    const int bh = blockIdx.y;
    const int b  = bh / H;
    const int h  = bh % H;
    const int q0 = (blockIdx.x * ATT_WAVES + wid) * 16;

    const float* Qb = q + (size_t)b * T * D + h * 64;
    const float* Kb = k + (size_t)b * T * D + h * 64;
    const float* Vb = v + (size_t)b * T * D + h * 64;
    float*       Ob = o + (size_t)b * T * D + h * 64;

    // Q fragments (A-layout), pre-scaled by 1/sqrt(HD)
    v16h qf[2];
    {
        const float* qrow = Qb + (size_t)(q0 + r) * D;
#pragma unroll
        for (int s = 0; s < 2; ++s)
            qf[s] = cvt8x2s(qrow + 32 * s + hl * 16, qrow + 32 * s + hl * 16 + 8, scale);
    }

    v8f oacc[4];
#pragma unroll
    for (int dt = 0; dt < 4; ++dt)
#pragma unroll
        for (int e = 0; e < 8; ++e) oacc[dt][e] = 0.0f;

    if (lane < 16) { mrow[wid][lane] = -INFINITY; lrow[wid][lane] = 0.0f; }

    const int nch = q0 / 32 + 1;  // causal: only key chunks with k0 <= q0+15
    for (int kc = 0; kc < nch; ++kc) {
        const int k0 = kc * 32;
        // --- scores S = (Q*scale) @ K^T for 32 keys -> two 16x16 tiles ---
#pragma unroll
        for (int jt = 0; jt < 2; ++jt) {
            v8f s;
#pragma unroll
            for (int e = 0; e < 8; ++e) s[e] = 0.0f;
#pragma unroll
            for (int st = 0; st < 2; ++st) {
                const float* krow =
                    Kb + (size_t)(k0 + jt * 16 + r) * D + 32 * st + hl * 16;
                v16h bfrag = cvt16(krow);
                s = WMMA16(qf[st], bfrag, s);
            }
#pragma unroll
            for (int e = 0; e < 8; ++e)
                Sld[wid][(e + 8 * hl) * 32 + jt * 16 + r] = s[e];
        }
        asm volatile("s_wait_dscnt 0" ::: "memory");

        // --- online softmax, one row per lane (lanes 0..15) ---
        if (lane < 16) {
            const int rr   = lane;
            const int qabs = q0 + rr;
            int vmax = qabs - k0;              // always >= 0 here
            if (vmax > 31) vmax = 31;
            const float mold = mrow[wid][rr];
            float mx = mold;
            for (int c = 0; c <= vmax; ++c) mx = fmaxf(mx, Sld[wid][rr * 32 + c]);
            const float alpha = __expf(mold - mx);  // exp(-inf)=0 on first chunk
            float ls = 0.0f;
            for (int c = 0; c < 32; ++c) {
                float p = (c <= vmax) ? __expf(Sld[wid][rr * 32 + c] - mx) : 0.0f;
                Pld[wid][rr * 32 + c] = (_Float16)p;
                ls += p;
            }
            mrow[wid][rr] = mx;
            lrow[wid][rr] = lrow[wid][rr] * alpha + ls;
            arow[wid][rr] = alpha;
        }
        asm volatile("s_wait_dscnt 0" ::: "memory");

        // rescale O accumulators by alpha (row = e + 8*hl)
        float al[8];
#pragma unroll
        for (int e = 0; e < 8; ++e) al[e] = arow[wid][e + 8 * hl];
#pragma unroll
        for (int dt = 0; dt < 4; ++dt)
#pragma unroll
            for (int e = 0; e < 8; ++e) oacc[dt][e] *= al[e];

        // P fragment (A-layout) from LDS
        v16h pf;
        {
            const _Float16* prow = &Pld[wid][r * 32];
#pragma unroll
            for (int i = 0; i < 8; ++i) {
                pf[i]     = prow[hl * 16 + i];
                pf[i + 8] = prow[hl * 16 + 8 + i];
            }
        }
        // O += P @ V  (4 tiles of 16 cols)
#pragma unroll
        for (int dt = 0; dt < 4; ++dt) {
            v16h vf;
#pragma unroll
            for (int i = 0; i < 16; ++i)
                vf[i] = (_Float16)Vb[(size_t)(k0 + hl * 16 + i) * D + dt * 16 + r];
            oacc[dt] = WMMA16(pf, vf, oacc[dt]);
        }
    }

    asm volatile("s_wait_dscnt 0" ::: "memory");
    float li[8];
#pragma unroll
    for (int e = 0; e < 8; ++e) li[e] = 1.0f / lrow[wid][e + 8 * hl];
#pragma unroll
    for (int dt = 0; dt < 4; ++dt)
#pragma unroll
        for (int e = 0; e < 8; ++e)
            Ob[(size_t)(q0 + e + 8 * hl) * D + dt * 16 + r] = oacc[dt][e] * li[e];
}

// ---------------------------------------------------------------------------
// RMSNorm: one block (256 threads) per row. Safe for in==out.
// ---------------------------------------------------------------------------
__global__ void __launch_bounds__(256)
rmsnorm_kernel(const float* __restrict__ in, float* __restrict__ out, int D) {
    __shared__ float red[8];
    __shared__ float sc_sh;
    const size_t row = blockIdx.x;
    const float* x = in + row * D;
    float s = 0.0f;
    for (int i = threadIdx.x; i < D; i += 256) { float vv = x[i]; s += vv * vv; }
#pragma unroll
    for (int off = 16; off > 0; off >>= 1) s += __shfl_xor(s, off, 32);
    const int lane = threadIdx.x & 31, w = threadIdx.x >> 5;
    if (lane == 0) red[w] = s;
    __syncthreads();
    if (threadIdx.x == 0) {
        float t = 0.0f;
        for (int i = 0; i < 8; ++i) t += red[i];
        sc_sh = rsqrtf(t / (float)D + 1e-5f);
    }
    __syncthreads();
    const float sc = sc_sh;
    for (int i = threadIdx.x; i < D; i += 256) out[row * D + i] = x[i] * sc;
}

// ---------------------------------------------------------------------------
// Embedding: x[m] = wte[idx[m]] + wpe[m % T]
// ---------------------------------------------------------------------------
__global__ void __launch_bounds__(256)
embed_kernel(const int* __restrict__ idx, const float* __restrict__ wte,
             const float* __restrict__ wpe, float* __restrict__ x, int T, int D) {
    const int m = blockIdx.x;
    const int t = m % T;
    const int id = idx[m];
    const float* te = wte + (size_t)id * D;
    const float* pe = wpe + (size_t)t * D;
    float* xr = x + (size_t)m * D;
    for (int i = threadIdx.x; i < D; i += 256) xr[i] = te[i] + pe[i];
}

// ---------------------------------------------------------------------------
// Gate softmax + top-2: one warp per token (E=8, D=1024).
// Writes normalized top-2 weights (zeros elsewhere) and accumulates prob sums.
// ---------------------------------------------------------------------------
__global__ void __launch_bounds__(256)
gate_topk_kernel(const float* __restrict__ xn, const float* __restrict__ gweight,
                 float* __restrict__ gw, float* __restrict__ probsum,
                 int D, int E) {
    const int lane = threadIdx.x & 31;
    const int warp = threadIdx.x >> 5;
    const int tok  = blockIdx.x * 8 + warp;
    const float* xrow = xn + (size_t)tok * D;
    float logit[8];
#pragma unroll
    for (int e = 0; e < 8; ++e) {
        const float* wrow = gweight + (size_t)e * D;
        float s = 0.0f;
        for (int i = lane; i < D; i += 32) s += xrow[i] * wrow[i];
#pragma unroll
        for (int off = 16; off > 0; off >>= 1) s += __shfl_xor(s, off, 32);
        logit[e] = s;
    }
    if (lane == 0) {
        float mx = logit[0];
        for (int e = 1; e < 8; ++e) mx = fmaxf(mx, logit[e]);
        float pe[8], sum = 0.0f;
        for (int e = 0; e < 8; ++e) { pe[e] = __expf(logit[e] - mx); sum += pe[e]; }
        const float inv = 1.0f / sum;
        for (int e = 0; e < 8; ++e) pe[e] *= inv;
        int i1 = 0;
        for (int e = 1; e < 8; ++e) if (pe[e] > pe[i1]) i1 = e;
        int i2 = -1;
        for (int e = 0; e < 8; ++e)
            if (e != i1 && (i2 < 0 || pe[e] > pe[i2])) i2 = e;
        const float wsum = pe[i1] + pe[i2];
        for (int e = 0; e < 8; ++e)
            gw[(size_t)tok * E + e] =
                (e == i1) ? pe[i1] / wsum : ((e == i2) ? pe[i2] / wsum : 0.0f);
        for (int e = 0; e < 8; ++e) atomicAdd(&probsum[e], pe[e]);
    }
}

__global__ void zero_kernel(float* p, int n) {
    const int i = blockIdx.x * blockDim.x + threadIdx.x;
    if (i < n) p[i] = 0.0f;
}

__global__ void aux_accum_kernel(const float* __restrict__ probsum,
                                 float* __restrict__ aux, float invMT, int E) {
    if (threadIdx.x == 0 && blockIdx.x == 0) {
        float s = 0.0f;
        for (int e = 0; e < E; ++e) { float f = probsum[e] * invMT; s += f * f; }
        *aux += (float)E * s;
    }
}

__global__ void aux_final_kernel(const float* __restrict__ aux,
                                 float* __restrict__ out, float coef) {
    if (threadIdx.x == 0 && blockIdx.x == 0) out[0] = coef * aux[0];
}

// ---------------------------------------------------------------------------
// Host-side orchestration
// ---------------------------------------------------------------------------
extern "C" void kernel_launch(void* const* d_in, const int* in_sizes, int n_in,
                              void* d_out, int out_size, void* d_ws, size_t ws_size,
                              hipStream_t stream) {
    (void)in_sizes; (void)n_in; (void)out_size; (void)ws_size;
    constexpr int B = 2, T = 1024, D = 1024, H = 16, L = 3, E = 8, FF = 4096, V = 32000;
    constexpr int M = B * T;  // 2048 tokens

    const int*   idx     = (const int*)d_in[0];
    const float* wte     = (const float*)d_in[1];
    const float* wpe     = (const float*)d_in[2];
    const float* wq      = (const float*)d_in[3];
    const float* wk      = (const float*)d_in[4];
    const float* wv      = (const float*)d_in[5];
    const float* wo      = (const float*)d_in[6];
    const float* gate    = (const float*)d_in[7];
    const float* f1      = (const float*)d_in[8];
    const float* f2      = (const float*)d_in[9];
    const float* lm_head = (const float*)d_in[10];
    float* out = (float*)d_out;

    float* ws = (float*)d_ws;
    float* x  = ws;                           // M*D
    float* xn = x  + (size_t)M * D;           // M*D
    float* qb = xn + (size_t)M * D;           // M*D
    float* kb = qb + (size_t)M * D;           // M*D
    float* vb = kb + (size_t)M * D;           // M*D
    float* ob = vb + (size_t)M * D;           // M*D
    float* hb = ob + (size_t)M * D;           // M*FF
    float* gw = hb + (size_t)M * FF;          // M*E
    float* probsum = gw + (size_t)M * E;      // E
    float* aux     = probsum + E;             // 1

    const dim3 blk(256);
    const dim3 gDD(D / 128, M / 128);   // N=D GEMMs
    const dim3 gDF(FF / 128, M / 128);  // N=FF GEMM
    const dim3 gDV(V / 128, M / 128);   // lm_head GEMM

    zero_kernel<<<1, 32, 0, stream>>>(aux, 1);
    embed_kernel<<<M, blk, 0, stream>>>(idx, wte, wpe, x, T, D);
    rmsnorm_kernel<<<M, blk, 0, stream>>>(x, x, D);

    for (int l = 0; l < L; ++l) {
        rmsnorm_kernel<<<M, blk, 0, stream>>>(x, xn, D);
        gemm_wmma_kernel<0, 0, 0><<<gDD, blk, 0, stream>>>(
            xn, wq + (size_t)l * D * D, qb, nullptr, nullptr, 0, M, D, D);
        gemm_wmma_kernel<0, 0, 0><<<gDD, blk, 0, stream>>>(
            xn, wk + (size_t)l * D * D, kb, nullptr, nullptr, 0, M, D, D);
        gemm_wmma_kernel<0, 0, 0><<<gDD, blk, 0, stream>>>(
            xn, wv + (size_t)l * D * D, vb, nullptr, nullptr, 0, M, D, D);

        flash_attn_kernel<<<dim3(T / 16 / ATT_WAVES, B * H), dim3(ATT_WAVES * 32),
                            0, stream>>>(qb, kb, vb, ob, T, D, H, 0.125f);

        // x = attn_out @ wo^T + x  (residual fused)
        gemm_wmma_kernel<0, 0, 0><<<gDD, blk, 0, stream>>>(
            ob, wo + (size_t)l * D * D, x, x, nullptr, 0, M, D, D);

        // MoE
        rmsnorm_kernel<<<M, blk, 0, stream>>>(x, xn, D);
        zero_kernel<<<1, 32, 0, stream>>>(probsum, E);
        gate_topk_kernel<<<M / 8, blk, 0, stream>>>(
            xn, gate + (size_t)l * E * D, gw, probsum, D, E);
        aux_accum_kernel<<<1, 32, 0, stream>>>(probsum, aux, 1.0f / (float)M, E);

        for (int e = 0; e < E; ++e) {
            // h = relu(xn @ f1[l,e]^T)
            gemm_wmma_kernel<1, 0, 0><<<gDF, blk, 0, stream>>>(
                xn, f1 + ((size_t)(l * E + e)) * FF * D, hb,
                nullptr, nullptr, 0, M, FF, D);
            // x += gate_w[:,e] * (h @ f2[l,e]^T)
            gemm_wmma_kernel<0, 1, 1><<<gDD, blk, 0, stream>>>(
                hb, f2 + ((size_t)(l * E + e)) * D * FF, x,
                nullptr, gw + e, E, M, D, FF);
        }
    }

    // logits = x @ lm_head^T  (directly into d_out)
    gemm_wmma_kernel<0, 0, 0><<<gDV, blk, 0, stream>>>(
        x, lm_head, out, nullptr, nullptr, 0, M, V, D);
    aux_final_kernel<<<1, 32, 0, stream>>>(aux, out + (size_t)M * V, 0.01f / (float)L);
}
